// GraphConvModel_51771535786035
// MI455X (gfx1250) — matrix operand order repaired and checked by reading, not compiled, not run on hardware
//
#include <hip/hip_runtime.h>

// ---------------------------------------------------------------------------
// GraphConv x2 + MLP for MI455X (gfx1250, wave32, WMMA).
// SpMM via f32 atomics (L2-resident working set), dense layers via
// v_wmma_f32_16x16x32_f16 with f32 accumulation.
// ---------------------------------------------------------------------------

typedef __attribute__((ext_vector_type(16))) _Float16 v16h_t;
typedef __attribute__((ext_vector_type(8)))  float    v8f_t;

#define N_NODES 50000
#define N_EDGES 800000
#define IN_FEATS 128
#define HIDDEN 128
#define DNN_H 256
#define N_CLASSES 40

static inline int cdiv_host(long long a, long long b) { return (int)((a + b - 1) / b); }

// --------------------------- utility kernels -------------------------------

__global__ void zero_f32_kernel(float* __restrict__ p, long long n) {
    long long i = (long long)blockIdx.x * blockDim.x + threadIdx.x;
    if (i < n) p[i] = 0.0f;
}

__global__ void degree_kernel(const int* __restrict__ src, const int* __restrict__ dst,
                              float* __restrict__ deg_out, float* __restrict__ deg_in, int nE) {
    int e = blockIdx.x * blockDim.x + threadIdx.x;
    if (e >= nE) return;
    atomicAdd(&deg_out[src[e]], 1.0f);
    atomicAdd(&deg_in[dst[e]],  1.0f);
}

__global__ void inv_sqrt_kernel(float* __restrict__ d, long long n) {
    long long i = (long long)blockIdx.x * blockDim.x + threadIdx.x;
    if (i < n) d[i] = rsqrtf(fmaxf(d[i], 1.0f));
}

// SpMM scatter: one lane handles 4 contiguous features of one edge.
// 32 lanes cover the 128-wide feature row.
__global__ void spmm_scatter_kernel(const float* __restrict__ x,
                                    const int* __restrict__ src, const int* __restrict__ dst,
                                    const float* __restrict__ inv_sqrt_out,
                                    float* __restrict__ agg, int nE) {
    long long t = (long long)blockIdx.x * blockDim.x + threadIdx.x;
    int e    = (int)(t >> 5);
    int lane = (int)(t & 31);
    if (e >= nE) return;
    int s = src[e];
    int d = dst[e];
    float sc = inv_sqrt_out[s];
    const float4* xr = (const float4*)(x + (size_t)s * 128);
    float4 v = xr[lane];
    float* o = agg + (size_t)d * 128 + lane * 4;
    atomicAdd(o + 0, v.x * sc);
    atomicAdd(o + 1, v.y * sc);
    atomicAdd(o + 2, v.z * sc);
    atomicAdd(o + 3, v.w * sc);
}

// ----------------------------- WMMA GEMM -----------------------------------
// out[M x NCOLS] = act( (A * rowScale?) [M x K] @ W [K x NCOLS] + bias )
// 4 waves / block; each wave owns a 16-row stripe and sweeps NTILES of 16 cols.
// A fragments (all K chunks) preloaded once per wave, fused f32->f16 + scale.
// ACT: 0 = ReLU, 1 = LeakyReLU(0.01)

template <int K, int NCOLS, int NTILES, int ACT, bool SCALE>
__global__ __launch_bounds__(128) void gemm_wmma_kernel(
    const float* __restrict__ A, const float* __restrict__ rowScale,
    const float* __restrict__ W, const float* __restrict__ bias,
    float* __restrict__ out, int M)
{
    const int lane    = threadIdx.x & 31;
    const int wave    = threadIdx.x >> 5;
    const int rowBase = blockIdx.x * 64 + wave * 16;
    if (rowBase >= M) return;                 // wave-uniform exit: EXEC stays all-1s

    const int  mrow  = rowBase + (lane & 15);
    const int  halfk = lane >> 4;
    const float* arow = A + (size_t)mrow * K;
    const float s = SCALE ? rowScale[mrow] : 1.0f;

    constexpr int KC = K / 32;
    v16h_t afrag[KC];
#pragma unroll
    for (int kc = 0; kc < KC; ++kc) {
        const int kBase = kc * 32 + 8 * halfk;          // A layout (ISA 7.12.2)
#pragma unroll
        for (int i = 0; i < 8; ++i) {
            const int kb = kBase + ((i < 4) ? 0 : 16) + 2 * (i & 3);
            afrag[kc][2 * i]     = (_Float16)(arow[kb]     * s);
            afrag[kc][2 * i + 1] = (_Float16)(arow[kb + 1] * s);
        }
    }

    const int ncol = lane & 15;
#pragma unroll 1
    for (int nt = 0; nt < NTILES; ++nt) {
        const int  n      = nt * 16 + ncol;
        const bool nvalid = (n < NCOLS);
        v8f_t acc = {};
#pragma unroll
        for (int kc = 0; kc < KC; ++kc) {
            v16h_t bfrag;
            const int kBase = kc * 32 + 8 * halfk;      // B layout (ISA 7.12.2)
#pragma unroll
            for (int i = 0; i < 8; ++i) {
                const int k = kBase + ((i < 4) ? 0 : 16) + 2 * (i & 3);
                bfrag[2 * i]     = nvalid ? (_Float16)W[(size_t)k       * NCOLS + n] : (_Float16)0.0f;
                bfrag[2 * i + 1] = nvalid ? (_Float16)W[(size_t)(k + 1) * NCOLS + n] : (_Float16)0.0f;
            }
            acc = __builtin_amdgcn_wmma_f32_16x16x32_f16(
                /*neg_a=*/false, afrag[kc], /*neg_b=*/false, bfrag,
                /*c_mod=*/(short)0, acc, /*reuse_a=*/false, /*reuse_b=*/false);
        }
        if (nvalid) {
            const float bv = bias[n];
#pragma unroll
            for (int r = 0; r < 8; ++r) {
                const int m = rowBase + r + 8 * halfk;  // C/D layout (ISA 7.12.2)
                float v = acc[r] + bv;
                v = (ACT == 0) ? fmaxf(v, 0.0f) : (v < 0.0f ? 0.01f * v : v);
                out[(size_t)m * NCOLS + n] = v;
            }
        }
    }
}

// ------------------------------- driver ------------------------------------

extern "C" void kernel_launch(void* const* d_in, const int* in_sizes, int n_in,
                              void* d_out, int out_size, void* d_ws, size_t ws_size,
                              hipStream_t stream) {
    const float* features = (const float*)d_in[0];
    const int*   src      = (const int*)  d_in[1];
    const int*   dst      = (const int*)  d_in[2];
    const float* W1       = (const float*)d_in[3];
    const float* b1       = (const float*)d_in[4];
    const float* W2       = (const float*)d_in[5];
    const float* b2       = (const float*)d_in[6];
    const float* Wd1      = (const float*)d_in[7];
    const float* bd1      = (const float*)d_in[8];
    const float* Wd2      = (const float*)d_in[9];
    const float* bd2      = (const float*)d_in[10];
    float* out = (float*)d_out;

    // workspace layout (floats):
    //   [inv_sqrt_out N][inv_sqrt_in N][agg N*128][h N*128][hd1 N*256]
    float* ws  = (float*)d_ws;
    float* iso = ws;                               // deg_out -> inv_sqrt_out
    float* isi = ws + N_NODES;                     // deg_in  -> inv_sqrt_in
    float* agg = ws + 2 * (size_t)N_NODES;         // N x 128
    float* hbf = agg + (size_t)N_NODES * 128;      // N x 128 (h1, then reused for h2)
    float* hd1 = hbf + (size_t)N_NODES * 128;      // N x 256

    const int TPB = 256;
    const long long nDeg  = 2LL * N_NODES;
    const long long nAgg  = (long long)N_NODES * 128;
    const long long nScat = (long long)N_EDGES * 32;
    const int gemmGrid = cdiv_host(N_NODES, 64);

    // degrees -> inv-sqrt norms
    zero_f32_kernel<<<cdiv_host(nDeg, TPB), TPB, 0, stream>>>(iso, nDeg);
    degree_kernel<<<cdiv_host(N_EDGES, TPB), TPB, 0, stream>>>(src, dst, iso, isi, N_EDGES);
    inv_sqrt_kernel<<<cdiv_host(nDeg, TPB), TPB, 0, stream>>>(iso, nDeg);

    // layer 1: SpMM(features) -> agg ; relu(agg * inv_in @ W1 + b1) -> hbf
    zero_f32_kernel<<<cdiv_host(nAgg, TPB), TPB, 0, stream>>>(agg, nAgg);
    spmm_scatter_kernel<<<cdiv_host(nScat, TPB), TPB, 0, stream>>>(features, src, dst, iso, agg, N_EDGES);
    gemm_wmma_kernel<128, 128, 8, 0, true><<<gemmGrid, 128, 0, stream>>>(agg, isi, W1, b1, hbf, N_NODES);

    // layer 2: SpMM(hbf) -> agg ; relu(agg * inv_in @ W2 + b2) -> hbf (reuse)
    zero_f32_kernel<<<cdiv_host(nAgg, TPB), TPB, 0, stream>>>(agg, nAgg);
    spmm_scatter_kernel<<<cdiv_host(nScat, TPB), TPB, 0, stream>>>(hbf, src, dst, iso, agg, N_EDGES);
    gemm_wmma_kernel<128, 128, 8, 0, true><<<gemmGrid, 128, 0, stream>>>(agg, isi, W2, b2, hbf, N_NODES);

    // DNN: leaky_relu(hbf @ Wd1 + bd1) -> hd1 ; leaky_relu(hd1 @ Wd2 + bd2) -> out
    gemm_wmma_kernel<128, 256, 16, 1, false><<<gemmGrid, 128, 0, stream>>>(hbf, nullptr, Wd1, bd1, hd1, N_NODES);
    gemm_wmma_kernel<256,  40,  3, 1, false><<<gemmGrid, 128, 0, stream>>>(hd1, nullptr, Wd2, bd2, out, N_NODES);
}
